// GRU_56530359550206
// MI455X (gfx1250) — compile-verified, matrix operand order checked by reading, not run on hardware
//
#include <hip/hip_runtime.h>

// ---------------------------------------------------------------------------
// GRU for MI455X (gfx1250, wave32, WMMA + TDM).
//   B=64, T=512, DIN=D=512.  Output: last hidden state [64,512] fp32.
//
// phase 0: pack W_z/W_r/W_h/U_z/U_r/U_h fp32 -> bf16 WMMA-tile layout
// phase 1: xz/xr/xh = Xt @ W_* + b_*   (big parallel bf16 WMMA GEMMs, fp32 out)
// phase 2: sequential recurrence; batch split across 4 independent workgroups
//          (16 batch rows each == one WMMA M tile), state in LDS, U from L2,
//          per-step gate inputs double-buffered into LDS by the Tensor Data
//          Mover (TENSORcnt) one step ahead of use.
// ---------------------------------------------------------------------------

typedef __bf16 v16bf __attribute__((ext_vector_type(16)));
typedef float  v8f   __attribute__((ext_vector_type(8)));
typedef unsigned int u32x4 __attribute__((ext_vector_type(4)));
typedef int          i32x4 __attribute__((ext_vector_type(4)));
typedef int          i32x8 __attribute__((ext_vector_type(8)));

struct BPair { uint4 lo, hi; };          // 32 bytes == v16bf
struct A16   { unsigned short v[16]; };  // 32 bytes == v16bf

#if defined(__HIP_DEVICE_COMPILE__) && __has_builtin(__builtin_amdgcn_tensor_load_to_lds)
#define GRU_TDM 1
#else
#define GRU_TDM 0
#endif

__device__ __forceinline__ unsigned short f2bf(float f) {
    unsigned u = __builtin_bit_cast(unsigned, f);
    unsigned r = u + 0x7fffu + ((u >> 16) & 1u);   // round-to-nearest-even
    return (unsigned short)(r >> 16);
}
__device__ __forceinline__ float bf2f(unsigned short h) {
    return __builtin_bit_cast(float, (unsigned)h << 16);
}
__device__ __forceinline__ float sigmoidf_fast(float x) {
    return 1.0f / (1.0f + __expf(-x));
}
__device__ __forceinline__ float tanhf_fast(float x) {
    float xc = fminf(fmaxf(x, -15.0f), 15.0f);
    float e  = __expf(2.0f * xc);
    return (e - 1.0f) / (e + 1.0f);
}

__device__ __forceinline__ v8f wmma_bf16(v16bf a, v16bf b, v8f c) {
    return __builtin_amdgcn_wmma_f32_16x16x32_bf16(
        /*neg_a=*/false, a, /*neg_b=*/false, b,
        /*c_mod=*/(short)0, c, /*reuse_a=*/false, /*reuse_b=*/false);
}

// Packed B tile: [nt][kt][lane][16] bf16, 512 bf16 per (nt,kt) tile.
// lane: n = nt*16 + (lane&15); K chunks {c0..c0+7, c0+16..c0+23}, c0=(lane>>4)*8.
__device__ __forceinline__ v16bf ldB(const unsigned short* __restrict__ W,
                                     int nt, int kt, int lane) {
    const unsigned short* p = W + (((size_t)(nt * 16 + kt)) << 9) + (lane << 4);
    BPair t;
    t.lo = *(const uint4*)p;
    t.hi = *(const uint4*)(p + 8);
    return __builtin_bit_cast(v16bf, t);
}

// A operand from an LDS row (bf16, pitch-aligned): two 16B chunks at k0, k0+16.
__device__ __forceinline__ v16bf ldA_lds(const unsigned short* rowp, int k0) {
    BPair t;
    t.lo = *(const uint4*)(rowp + k0);
    t.hi = *(const uint4*)(rowp + k0 + 16);
    return __builtin_bit_cast(v16bf, t);
}

#if GRU_TDM
// TDM: DMA `nelem` contiguous 4-byte elements from global -> LDS (1-row tile).
// D# built per CDNA5 ISA (group0: count=1,type=2; group1:
// data_size=4B, tensor_dim0=tile_dim0=nelem, dims1..4 trivial). Tracked with
// TENSORcnt; caller pairs with s_wait_tensorcnt.
__device__ __forceinline__ void tdm_load_1d(const void* gptr, unsigned lds_addr,
                                            unsigned nelem) {
    unsigned long long ga = (unsigned long long)(size_t)gptr;
    u32x4 g0;
    g0[0] = 1u;                                           // count=1 (valid, user)
    g0[1] = lds_addr;                                     // LDS byte address
    g0[2] = (unsigned)(ga & 0xffffffffu);                 // global_addr[31:0]
    g0[3] = (unsigned)((ga >> 32) & 0x01ffffffu) | 0x80000000u;  // ga[56:32]|type=2
    i32x8 g1;
    g1[0] = 0x20000;                                      // data_size=2 (4 bytes)
    g1[1] = (int)((nelem & 0xffffu) << 16);               // tensor_dim0[15:0]
    g1[2] = (int)((nelem >> 16) | (1u << 16));            // tdim0[31:16] | tdim1=1
    g1[3] = (int)((nelem & 0xffffu) << 16);               // tile_dim0 (<=65535)
    g1[4] = 1;                                            // tile_dim1=1, tile_dim2=0
    g1[5] = (int)nelem;                                   // tdim0_stride[31:0]
    g1[6] = (int)(((nelem >> 16) & 0xffffu) | ((nelem & 0xffffu) << 16)); // s0 hi|s1 lo
    g1[7] = (int)((nelem >> 16) & 0xffffu);               // tdim1_stride[47:16]
    i32x4 z4 = {0, 0, 0, 0};
#if defined(__clang_major__) && __clang_major__ >= 23
    i32x8 z8 = {0, 0, 0, 0, 0, 0, 0, 0};
    __builtin_amdgcn_tensor_load_to_lds(g0, g1, z4, z4, z8, 0);
#else
    __builtin_amdgcn_tensor_load_to_lds(g0, g1, z4, z4, 0);
#endif
}
#endif  // GRU_TDM

// ---------------------------------------------------------------------------
// Phase 0: pack 6 fp32 [512,512] matrices into bf16 WMMA-tile layout.
// mats: 0=W_z 1=W_r 2=W_h 3=U_z 4=U_r 5=U_h ; each packed mat = 262144 ushort.
// ---------------------------------------------------------------------------
__global__ __launch_bounds__(256) void gru_pack_kernel(
    const float* __restrict__ Wz, const float* __restrict__ Wr,
    const float* __restrict__ Wh, const float* __restrict__ Uz,
    const float* __restrict__ Ur, const float* __restrict__ Uh,
    unsigned short* __restrict__ pk)
{
    int gid  = blockIdx.x * 256 + threadIdx.x;   // 0 .. 6*262144-1
    int mat  = gid >> 18;
    int rem  = gid & 262143;
    int i    = rem & 15;
    int lane = (rem >> 4) & 31;
    int kt   = (rem >> 9) & 15;
    int nt   = rem >> 13;
    const float* s = (mat == 0) ? Wz : (mat == 1) ? Wr : (mat == 2) ? Wh
                   : (mat == 3) ? Uz : (mat == 4) ? Ur : Uh;
    int n  = nt * 16 + (lane & 15);
    int c0 = (lane >> 4) * 8;
    int kl = (i < 8) ? (c0 + i) : (c0 + 16 + (i - 8));
    int k  = kt * 32 + kl;
    pk[gid] = f2bf(s[k * 512 + n]);
}

// ---------------------------------------------------------------------------
// Phase 1: proj[g][t*64+b][n] = sum_k X[b][t][k] * W_g[k][n] + b_g[n]
// grid (2048 Mtiles, 3 gates) x 256 threads (8 waves, 4 N-tiles each).
// ---------------------------------------------------------------------------
__global__ __launch_bounds__(256) void gru_proj_kernel(
    const float* __restrict__ X, const unsigned short* __restrict__ pk,
    const float* __restrict__ bz, const float* __restrict__ br,
    const float* __restrict__ bh, float* __restrict__ proj)
{
    const int mtile = blockIdx.x;
    const int g     = blockIdx.y;
    const unsigned short* W = pk + (size_t)g * 262144;
    const float* bias = (g == 0) ? bz : (g == 1) ? br : bh;
    float* out = proj + (size_t)g * 16777216;   // 64*512*512 floats per gate

    const int lane = threadIdx.x & 31;
    const int w    = threadIdx.x >> 5;
    const int mrow = lane & 15;
    const int c0   = (lane >> 4) * 8;
    const int r    = mtile * 16 + mrow;         // flat M row = t*64 + b
    const int tt   = r >> 6;
    const int bb   = r & 63;
    const float* xrow = X + ((size_t)bb * 512 + tt) * 512;

    v8f acc0 = {}, acc1 = {}, acc2 = {}, acc3 = {};
    for (int kt = 0; kt < 16; ++kt) {
        const int k0 = kt * 32 + c0;
        float4 x0 = *(const float4*)(xrow + k0);
        float4 x1 = *(const float4*)(xrow + k0 + 4);
        float4 x2 = *(const float4*)(xrow + k0 + 16);
        float4 x3 = *(const float4*)(xrow + k0 + 20);
        A16 at;
        at.v[0]  = f2bf(x0.x); at.v[1]  = f2bf(x0.y);
        at.v[2]  = f2bf(x0.z); at.v[3]  = f2bf(x0.w);
        at.v[4]  = f2bf(x1.x); at.v[5]  = f2bf(x1.y);
        at.v[6]  = f2bf(x1.z); at.v[7]  = f2bf(x1.w);
        at.v[8]  = f2bf(x2.x); at.v[9]  = f2bf(x2.y);
        at.v[10] = f2bf(x2.z); at.v[11] = f2bf(x2.w);
        at.v[12] = f2bf(x3.x); at.v[13] = f2bf(x3.y);
        at.v[14] = f2bf(x3.z); at.v[15] = f2bf(x3.w);
        v16bf a = __builtin_bit_cast(v16bf, at);
        acc0 = wmma_bf16(a, ldB(W, w,      kt, lane), acc0);
        acc1 = wmma_bf16(a, ldB(W, w + 8,  kt, lane), acc1);
        acc2 = wmma_bf16(a, ldB(W, w + 16, kt, lane), acc2);
        acc3 = wmma_bf16(a, ldB(W, w + 24, kt, lane), acc3);
    }

    const int ncol    = lane & 15;
    const int rowbase = 8 * (lane >> 4);
    #pragma unroll
    for (int q = 0; q < 4; ++q) {
        v8f acc = (q == 0) ? acc0 : (q == 1) ? acc1 : (q == 2) ? acc2 : acc3;
        const int nt = w + 8 * q;
        const int n  = nt * 16 + ncol;
        const float bn = bias[n];
        #pragma unroll
        for (int i = 0; i < 8; ++i) {
            const int rr = mtile * 16 + rowbase + i;  // C layout: M = i + 8*(lane>=16)
            out[(size_t)rr * 512 + n] = acc[i] + bn;
        }
    }
}

// ---------------------------------------------------------------------------
// Phase 2: recurrence. 4 workgroups (16 batch rows each), 512 threads (16 waves).
// Wave w owns output columns [32w, 32w+32) for z, r and the candidate.
// Dynamic LDS (floats): [0,24576) x-slices phase0 (z|r|h, 8192 each)
//                       [24576,49152) x-slices phase1
//                       [49152,57344) mask table 16x512
// ---------------------------------------------------------------------------
__global__ __launch_bounds__(512) void gru_rec_kernel(
    const float* __restrict__ xz, const float* __restrict__ xr,
    const float* __restrict__ xh, const unsigned short* __restrict__ pk,
    const int* __restrict__ mask, float* __restrict__ out)
{
    __shared__ __align__(16) float          h_f[16][520];   // fp32 hidden state
    __shared__ __align__(16) unsigned short hmrh[16][520];  // bf16: m*h then r*(m*h)
    extern __shared__ float dynx[];

    const unsigned short* Uz = pk + 3ull * 262144;
    const unsigned short* Ur = pk + 4ull * 262144;
    const unsigned short* Uh = pk + 5ull * 262144;

    const int mb0  = blockIdx.x * 16;
    const int tid  = threadIdx.x;
    const int lane = tid & 31;
    const int w    = tid >> 5;
    const int mrow = lane & 15;
    const int c0   = (lane >> 4) * 8;
    const int ntA  = 2 * w,      ntB  = 2 * w + 1;
    const int n0a  = ntA * 16,   n0b  = ntB * 16;
    const int ncol    = lane & 15;
    const int rowbase = 8 * (lane >> 4);
    const int col = w * 32 + lane;   // this thread's column for state maintenance

    float* mask_f = dynx + 49152;

    for (int idx = tid; idx < 16 * 520; idx += 512) (&h_f[0][0])[idx] = 0.0f;
    for (int idx = tid; idx < 8192; idx += 512)
        mask_f[idx] = (float)mask[(mb0 + (idx >> 9)) * 512 + (idx & 511)];

#if GRU_TDM
    const unsigned dyn_base = (unsigned)__builtin_amdgcn_groupstaticsize();
    if (w < 3) {   // waves 0..2: DMA step-0 gate inputs (uniform branch)
        const float* src = (w == 0) ? xz : (w == 1) ? xr : xh;
        tdm_load_1d(src + (size_t)mb0 * 512, dyn_base + (unsigned)(w * 32768), 8192);
        __builtin_amdgcn_s_wait_tensorcnt(0);
    }
#endif
    __syncthreads();

    const unsigned short* hrow = &hmrh[mrow][0];

    for (int t = 0; t < 512; ++t) {
#if GRU_TDM
        // prefetch next step's gate inputs into the other LDS phase
        if (t + 1 < 512 && w < 3) {
            const float* src = (w == 0) ? xz : (w == 1) ? xr : xh;
            tdm_load_1d(src + ((size_t)(t + 1) * 64 + mb0) * 512,
                        dyn_base + (unsigned)((((t + 1) & 1) * 98304) + w * 32768),
                        8192);
        }
        const float* xzb = dynx + (t & 1) * 24576;
        const float* xrb = xzb + 8192;
        const float* xhb = xzb + 16384;
#endif
        // ---- hm = m_tm1 * h_{t-1}  (bf16, full [16,512]) ----
        #pragma unroll 4
        for (int r2 = 0; r2 < 16; ++r2) {
            float m = (t > 0) ? mask_f[r2 * 512 + (t - 1)] : 0.0f;
            hmrh[r2][col] = f2bf(m * h_f[r2][col]);
        }
        __syncthreads();

        // ---- z/r GEMMs: [16,512] @ [512, 32 own cols] each ----
        v8f az0 = {}, az1 = {}, ar0 = {}, ar1 = {};
        for (int kt = 0; kt < 16; ++kt) {
            const int k0 = kt * 32 + c0;
            v16bf a = ldA_lds(hrow, k0);
            az0 = wmma_bf16(a, ldB(Uz, ntA, kt, lane), az0);
            az1 = wmma_bf16(a, ldB(Uz, ntB, kt, lane), az1);
            ar0 = wmma_bf16(a, ldB(Ur, ntA, kt, lane), ar0);
            ar1 = wmma_bf16(a, ldB(Ur, ntB, kt, lane), ar1);
            // warm the U_h stream for the candidate GEMM
            __builtin_prefetch((const void*)(Uh + (((size_t)(ntA * 16 + kt)) << 9)
                                             + (lane << 4)), 0, 1);
        }

        float zv[16], rv[16];
        #pragma unroll
        for (int i = 0; i < 8; ++i) {
            const int row = rowbase + i;
#if GRU_TDM
            zv[i]     = sigmoidf_fast(az0[i] + xzb[row * 512 + n0a + ncol]);
            zv[8 + i] = sigmoidf_fast(az1[i] + xzb[row * 512 + n0b + ncol]);
            rv[i]     = sigmoidf_fast(ar0[i] + xrb[row * 512 + n0a + ncol]);
            rv[8 + i] = sigmoidf_fast(ar1[i] + xrb[row * 512 + n0b + ncol]);
#else
            const size_t base = ((size_t)t * 64 + mb0 + row) * 512;
            zv[i]     = sigmoidf_fast(az0[i] + xz[base + n0a + ncol]);
            zv[8 + i] = sigmoidf_fast(az1[i] + xz[base + n0b + ncol]);
            rv[i]     = sigmoidf_fast(ar0[i] + xr[base + n0a + ncol]);
            rv[8 + i] = sigmoidf_fast(ar1[i] + xr[base + n0b + ncol]);
#endif
        }
        __syncthreads();   // all GEMM A-reads of hm done before overwrite

        // ---- overwrite buffer in place with r * hm; keep hm in regs ----
        float hmk[16];
        #pragma unroll
        for (int i = 0; i < 8; ++i) {
            const int row = rowbase + i;
            float hm0 = bf2f(hmrh[row][n0a + ncol]);
            float hm1 = bf2f(hmrh[row][n0b + ncol]);
            hmk[i]     = hm0;
            hmk[8 + i] = hm1;
            hmrh[row][n0a + ncol] = f2bf(rv[i]     * hm0);
            hmrh[row][n0b + ncol] = f2bf(rv[8 + i] * hm1);
        }
        __syncthreads();

        // ---- candidate GEMM: (r*hm) @ U_h ----
        v8f ah0 = {}, ah1 = {};
        for (int kt = 0; kt < 16; ++kt) {
            const int k0 = kt * 32 + c0;
            v16bf a = ldA_lds(hrow, k0);
            ah0 = wmma_bf16(a, ldB(Uh, ntA, kt, lane), ah0);
            ah1 = wmma_bf16(a, ldB(Uh, ntB, kt, lane), ah1);
        }

        // ---- h_t = z*hm + (1-z)*tanh(xh + rh@U_h) ----
        #pragma unroll
        for (int i = 0; i < 8; ++i) {
            const int row = rowbase + i;
#if GRU_TDM
            float hh0 = tanhf_fast(ah0[i] + xhb[row * 512 + n0a + ncol]);
            float hh1 = tanhf_fast(ah1[i] + xhb[row * 512 + n0b + ncol]);
#else
            const size_t base = ((size_t)t * 64 + mb0 + row) * 512;
            float hh0 = tanhf_fast(ah0[i] + xh[base + n0a + ncol]);
            float hh1 = tanhf_fast(ah1[i] + xh[base + n0b + ncol]);
#endif
            h_f[row][n0a + ncol] = zv[i]     * hmk[i]     + (1.0f - zv[i])     * hh0;
            h_f[row][n0b + ncol] = zv[8 + i] * hmk[8 + i] + (1.0f - zv[8 + i]) * hh1;
        }

#if GRU_TDM
        // publish next step's DMA before releasing the step barrier
        if (w < 3) __builtin_amdgcn_s_wait_tensorcnt(0);
#endif
        __syncthreads();   // h_f/hmrh/x-buffers stable before next step
    }

    // ---- emit last hidden state ----
    #pragma unroll 4
    for (int r2 = 0; r2 < 16; ++r2)
        out[(size_t)(mb0 + r2) * 512 + col] = h_f[r2][col];
}

// ---------------------------------------------------------------------------
extern "C" void kernel_launch(void* const* d_in, const int* in_sizes, int n_in,
                              void* d_out, int out_size, void* d_ws, size_t ws_size,
                              hipStream_t stream) {
    (void)in_sizes; (void)n_in; (void)out_size; (void)ws_size;
    const float* X   = (const float*)d_in[0];
    const float* W_z = (const float*)d_in[1];
    const float* U_z = (const float*)d_in[2];
    const float* b_z = (const float*)d_in[3];
    const float* W_r = (const float*)d_in[4];
    const float* U_r = (const float*)d_in[5];
    const float* b_r = (const float*)d_in[6];
    const float* W_h = (const float*)d_in[7];
    const float* U_h = (const float*)d_in[8];
    const float* b_h = (const float*)d_in[9];
    const int*  mask = (const int*)d_in[10];

    // workspace layout: [packed bf16 weights: 6*262144 ushort = 3 MiB]
    //                   [xz|xr|xh: 3 * 16777216 fp32 = 192 MiB]
    unsigned short* pk = (unsigned short*)d_ws;
    float* proj = (float*)((char*)d_ws + 6ull * 262144 * sizeof(unsigned short));

    gru_pack_kernel<<<6144, 256, 0, stream>>>(W_z, W_r, W_h, U_z, U_r, U_h, pk);
    gru_proj_kernel<<<dim3(2048, 3, 1), 256, 0, stream>>>(X, pk, b_z, b_r, b_h, proj);
    // dynamic LDS: 2*3*32KB x double-buffer + 32KB mask table = 229376 bytes
    gru_rec_kernel<<<4, 512, 229376, stream>>>(proj,
                                               proj + 16777216ull,
                                               proj + 2ull * 16777216ull,
                                               pk, mask, (float*)d_out);
}
